// DynamicGraphTransformerNetwork_29231547417199
// MI455X (gfx1250) — compile-verified, matrix operand order checked by reading, not run on hardware
//
#include <hip/hip_runtime.h>
#include <hip/hip_bf16.h>
#include <math.h>

// ---------------------------------------------------------------------------
// DynamicGraphTransformerNetwork on MI455X (gfx1250, wave32, WMMA).
//
// Problem sizes: B=1024, N=100, H=128, NH=8, DH=16, DFF=256, NL=3, T=200.
// M = B*N = 102400 rows for all big GEMMs (multiple of 16 -> no tile padding).
//
// Strategy:
//  * All dense GEMMs run through v_wmma_f32_16x16x32_f16 (fp16 in, fp32 acc).
//    Weights are converted+transposed to fp16 once per launch so weight
//    fragments are two global_load_b128 per 32-wide K chunk.
//  * Each block owns one 16-row A tile, staged once into LDS (pitch K+8 halfs
//    to stagger the 64 banks); its 8 waves cover 8 column tiles and read A
//    fragments via ds_load_b128 -> no redundant global A traffic.
//  * Decoder algebra is refactored so the step-invariant halves of the concat
//    GEMMs (emb@dyn_w1[:H], emb-part of ptr_w1) are hoisted out of the loop.
//    Per step only two [102400,128]x[128,128] WMMA GEMMs remain, with fused
//    epilogues (relu / ue=emb+0.1*upd / score row-dot via shfl + atomicAdd).
// ---------------------------------------------------------------------------

#define B_   1024
#define N_   100
#define H_   128
#define NH_  8
#define DH_  16
#define DFF_ 256
#define NL_  3
#define MAXS 200
#define M_   (B_ * N_)          // 102400
#define NEGV (-1e9f)
#define LNEPS 1e-5f

typedef _Float16 v8h  __attribute__((ext_vector_type(8)));
typedef _Float16 v16h __attribute__((ext_vector_type(16)));
typedef float    v8f  __attribute__((ext_vector_type(8)));
typedef float    v4f  __attribute__((ext_vector_type(4)));

// ---------------------------------------------------------------------------
// Fragment loaders for V_WMMA_F32_16X16X32_F16 (wave32).
// Lanes 0..15 hold K = kc+{0..7,16..23}; lanes 16..31 hold K = kc+{8..15,24..31};
// each 8-half run is one 16-byte load.
// ---------------------------------------------------------------------------
__device__ __forceinline__ v16h load_frag(const _Float16* __restrict__ base,
                                          int kc, int lane) {
  int kb = kc + ((lane >> 4) << 3);
  v8h lo = *(const v8h*)(base + kb);
  v8h hi = *(const v8h*)(base + kb + 16);
  v16h r;
#pragma unroll
  for (int i = 0; i < 8; ++i) { r[i] = lo[i]; r[i + 8] = hi[i]; }
  return r;
}

__device__ __forceinline__ v16h combine8(v8h lo, v8h hi) {
  v16h r;
#pragma unroll
  for (int i = 0; i < 8; ++i) { r[i] = lo[i]; r[i + 8] = hi[i]; }
  return r;
}

// ---------------------------------------------------------------------------
// Generic tiled WMMA GEMM:  C[M,Nc] = act(A[M,K](f16) @ W[K,Nc] + bias)
// Wt fp16 transposed: Wt[n*K + k]. Block = 256 thr = 8 waves = one 16-row A
// tile x 8 column tiles; A staged through LDS.
// ---------------------------------------------------------------------------
template <int RELU, int OUT16>
__global__ void gemm_f16(const _Float16* __restrict__ A,
                         const _Float16* __restrict__ Wt,
                         const float* __restrict__ bias,
                         float* __restrict__ Cf, _Float16* __restrict__ Ch,
                         int M, int K, int Nc) {
  __shared__ _Float16 As[16 * 264];              // K<=256, pitch=K+8
  const int pitch = K + 8;
  int tiles_n = Nc >> 4;
  int groups  = tiles_n >> 3;                    // 1 (Nc=128) or 2 (Nc=256)
  int tm  = blockIdx.x / groups;
  int grp = blockIdx.x - tm * groups;
  int segs = K >> 3;                             // 8-half segments per row
  for (int e = threadIdx.x; e < 16 * segs; e += blockDim.x) {
    int row = e / segs, seg = e - row * segs;
    *(v8h*)(As + row * pitch + seg * 8) =
        *(const v8h*)(A + (size_t)(tm * 16 + row) * K + seg * 8);
  }
  __syncthreads();
  int wave = threadIdx.x >> 5, lane = threadIdx.x & 31;
  int tn = grp * 8 + wave;
  const _Float16* Asr  = As + (lane & 15) * pitch;
  const _Float16* Wcol = Wt + (size_t)(tn * 16 + (lane & 15)) * K;
  v8f acc = {};
  for (int kc = 0; kc < K; kc += 32) {
    int kb = kc + ((lane >> 4) << 3);
    v16h a = combine8(*(const v8h*)(Asr + kb), *(const v8h*)(Asr + kb + 16));
    v16h b = load_frag(Wcol, kc, lane);
    acc = __builtin_amdgcn_wmma_f32_16x16x32_f16(false, a, false, b,
                                                 (short)0, acc, false, false);
  }
  int col   = tn * 16 + (lane & 15);
  int rbase = tm * 16 + ((lane >> 4) << 3);
  float bs = bias ? bias[col] : 0.f;
#pragma unroll
  for (int v = 0; v < 8; ++v) {
    float x = acc[v] + bs;
    if (RELU) x = fmaxf(x, 0.f);
    size_t o = (size_t)(rbase + v) * Nc + col;
    if (OUT16) Ch[o] = (_Float16)x; else Cf[o] = x;
  }
}

// ---------------------------------------------------------------------------
// Per-step GEMM 1:  upd = relu(emb_dyn + se_dyn[b]) @ dyn_w2 + b2
//                   ue  = emb + 0.1*upd        (fp16 out, feeds GEMM 2)
// The relu(add) + f16 convert is done ONCE per tile during the cooperative
// LDS stage (float4 vector loads), not per wave.
// ---------------------------------------------------------------------------
__global__ void gemm1_step(const float* __restrict__ embDyn,
                           const float* __restrict__ seDyn,
                           const _Float16* __restrict__ Wt,   // dyn_w2^T fp16
                           const float* __restrict__ b2,
                           const float* __restrict__ embF,
                           _Float16* __restrict__ ue) {
  __shared__ _Float16 As[16 * 136];              // K=128, pitch=136
  const int pitch = 136;
  int tm = blockIdx.x;
  {
    int e = threadIdx.x;                         // 256 thr = 16 rows x 16 segs
    int row = e >> 4, seg = e & 15;
    int m  = tm * 16 + row;
    int bb = m / N_;
    const float* er = embDyn + (size_t)m * H_ + seg * 8;
    const float* sr = seDyn  + (size_t)bb * H_ + seg * 8;
    v4f e0 = *(const v4f*)er,       e1 = *(const v4f*)(er + 4);
    v4f s0 = *(const v4f*)sr,       s1 = *(const v4f*)(sr + 4);
    v8h t;
#pragma unroll
    for (int i = 0; i < 4; ++i) {
      t[i]     = (_Float16)fmaxf(e0[i] + s0[i], 0.f);
      t[i + 4] = (_Float16)fmaxf(e1[i] + s1[i], 0.f);
    }
    *(v8h*)(As + row * pitch + seg * 8) = t;
  }
  __syncthreads();
  int wave = threadIdx.x >> 5, lane = threadIdx.x & 31;
  int tn = wave;
  const _Float16* Asr  = As + (lane & 15) * pitch;
  const _Float16* Wcol = Wt + (size_t)(tn * 16 + (lane & 15)) * H_;
  v8f acc = {};
  for (int kc = 0; kc < H_; kc += 32) {
    int kb = kc + ((lane >> 4) << 3);
    v16h a = combine8(*(const v8h*)(Asr + kb), *(const v8h*)(Asr + kb + 16));
    v16h b = load_frag(Wcol, kc, lane);
    acc = __builtin_amdgcn_wmma_f32_16x16x32_f16(false, a, false, b,
                                                 (short)0, acc, false, false);
  }
  int col   = tn * 16 + (lane & 15);
  int rbase = tm * 16 + ((lane >> 4) << 3);
  float bs = b2[col];
#pragma unroll
  for (int v = 0; v < 8; ++v) {
    size_t o = (size_t)(rbase + v) * H_ + col;
    ue[o] = (_Float16)(embF[o] + 0.1f * (acc[v] + bs));
  }
}

// ---------------------------------------------------------------------------
// Per-step GEMM 2:  h = relu(ue @ ptrA + ctxsum[b]);  scores += h . ptr_w2
// Row-dot over the tile's 16 columns via shfl-xor; cross-tile partials via
// global_atomic_add_f32 into scores (pre-seeded with ptr_b2).
// ---------------------------------------------------------------------------
__global__ void gemm2_step(const _Float16* __restrict__ ue,
                           const _Float16* __restrict__ Wt,   // ptr_w1[:H]^T fp16
                           const float* __restrict__ ctxsum,
                           const float* __restrict__ pw2,
                           float* __restrict__ scores) {
  __shared__ _Float16 As[16 * 136];
  const int pitch = 136;
  int tm = blockIdx.x;
  {
    int e = threadIdx.x;
    int row = e >> 4, seg = e & 15;
    *(v8h*)(As + row * pitch + seg * 8) =
        *(const v8h*)(ue + (size_t)(tm * 16 + row) * H_ + seg * 8);
  }
  __syncthreads();
  int wave = threadIdx.x >> 5, lane = threadIdx.x & 31;
  int tn = wave;
  const _Float16* Asr  = As + (lane & 15) * pitch;
  const _Float16* Wcol = Wt + (size_t)(tn * 16 + (lane & 15)) * H_;
  v8f acc = {};
  for (int kc = 0; kc < H_; kc += 32) {
    int kb = kc + ((lane >> 4) << 3);
    v16h a = combine8(*(const v8h*)(Asr + kb), *(const v8h*)(Asr + kb + 16));
    v16h b = load_frag(Wcol, kc, lane);
    acc = __builtin_amdgcn_wmma_f32_16x16x32_f16(false, a, false, b,
                                                 (short)0, acc, false, false);
  }
  int col   = tn * 16 + (lane & 15);
  int rbase = tm * 16 + ((lane >> 4) << 3);
  float w2c = pw2[col];
#pragma unroll
  for (int v = 0; v < 8; ++v) {
    int row = rbase + v;
    int bb  = row / N_;
    float h = fmaxf(acc[v] + ctxsum[(size_t)bb * H_ + col], 0.f);
    float part = h * w2c;
#pragma unroll
    for (int off = 1; off < 16; off <<= 1) part += __shfl_xor(part, off, 32);
    if ((lane & 15) == 0) atomicAdd(&scores[row], part);
  }
}

// ---------------------------------------------------------------------------
// Weight convert + transpose: src fp32 [K,Nc] (rows k0.., leading dim Nc)
// -> dst fp16 [Nc,K] ("column-major" for fragment loads).
// ---------------------------------------------------------------------------
__global__ void conv_w_f16t(const float* __restrict__ src,
                            _Float16* __restrict__ dst, int K, int Nc, int k0) {
  int idx = blockIdx.x * 256 + threadIdx.x;
  if (idx >= K * Nc) return;
  int k = idx / Nc, n = idx % Nc;
  dst[(size_t)n * K + k] = (_Float16)src[(size_t)(k0 + k) * Nc + n];
}

// Embedding: x = [cx, cy, demand] @ embed_w + embed_b  (fp32 + fp16 copy).
__global__ void embed_k(const float* __restrict__ coords,
                        const float* __restrict__ demands,
                        const float* __restrict__ w, const float* __restrict__ bias,
                        float* __restrict__ xf, _Float16* __restrict__ xh) {
  int r = blockIdx.x, j = threadIdx.x;
  float f0 = coords[(size_t)r * 2], f1 = coords[(size_t)r * 2 + 1], f2 = demands[r];
  float y = bias[j] + f0 * w[j] + f1 * w[H_ + j] + f2 * w[2 * H_ + j];
  xf[(size_t)r * H_ + j] = y;
  xh[(size_t)r * H_ + j] = (_Float16)y;
}

// Per-head attention: softmax(q k^T / sqrt(DH)) v. One block per (b, head).
__global__ void attn_k(const _Float16* __restrict__ q,
                       const _Float16* __restrict__ k,
                       const _Float16* __restrict__ v,
                       _Float16* __restrict__ out) {
  int b = blockIdx.x >> 3, hd = blockIdx.x & 7;
  __shared__ _Float16 Ks[N_ * DH_], Vs[N_ * DH_];
  for (int e = threadIdx.x; e < N_ * DH_; e += blockDim.x) {
    int n = e >> 4, d = e & 15;
    size_t src = ((size_t)b * N_ + n) * H_ + hd * DH_ + d;
    Ks[e] = k[src]; Vs[e] = v[src];
  }
  __syncthreads();
  int i = threadIdx.x;
  if (i < N_) {
    float qi[DH_];
    size_t qoff = ((size_t)b * N_ + i) * H_ + hd * DH_;
#pragma unroll
    for (int d = 0; d < DH_; ++d) qi[d] = (float)q[qoff + d];
    float mx = -1e30f;
    for (int jn = 0; jn < N_; ++jn) {
      float s = 0.f;
#pragma unroll
      for (int d = 0; d < DH_; ++d) s += qi[d] * (float)Ks[jn * DH_ + d];
      mx = fmaxf(mx, s * 0.25f);
    }
    float se = 0.f, o[DH_] = {};
    for (int jn = 0; jn < N_; ++jn) {
      float s = 0.f;
#pragma unroll
      for (int d = 0; d < DH_; ++d) s += qi[d] * (float)Ks[jn * DH_ + d];
      float w = __expf(s * 0.25f - mx);
      se += w;
#pragma unroll
      for (int d = 0; d < DH_; ++d) o[d] += w * (float)Vs[jn * DH_ + d];
    }
    float inv = 1.f / se;
#pragma unroll
    for (int d = 0; d < DH_; ++d) out[qoff + d] = (_Float16)(o[d] * inv);
  }
}

// y = LayerNorm(a + res) * s + b ; fp32 + fp16 copies. One block per row.
__global__ void add_ln(const float* __restrict__ a, const float* __restrict__ res,
                       const float* __restrict__ s, const float* __restrict__ bias,
                       float* __restrict__ outf, _Float16* __restrict__ outh) {
  int r = blockIdx.x, j = threadIdx.x;
  __shared__ float red[H_];
  float v = a[(size_t)r * H_ + j] + res[(size_t)r * H_ + j];
  red[j] = v; __syncthreads();
  for (int o = 64; o > 0; o >>= 1) { if (j < o) red[j] += red[j + o]; __syncthreads(); }
  float mean = red[0] * (1.f / H_); __syncthreads();
  float d = v - mean;
  red[j] = d * d; __syncthreads();
  for (int o = 64; o > 0; o >>= 1) { if (j < o) red[j] += red[j + o]; __syncthreads(); }
  float var = red[0] * (1.f / H_);
  float y = s[j] * d * rsqrtf(var + LNEPS) + bias[j];
  outf[(size_t)r * H_ + j] = y;
  outh[(size_t)r * H_ + j] = (_Float16)y;
}

// Reset decoder state every launch (deterministic, graph-safe).
__global__ void init_state(unsigned char* visited, int* cur, float* rem,
                           int* doneF, float* lps, const float* caps) {
  int i = blockIdx.x * 256 + threadIdx.x;
  if (i < B_ * N_) visited[i] = 0;
  if (i < B_) { cur[i] = 0; rem[i] = caps[i]; doneF[i] = 0; lps[i] = 0.f; }
}

// Step prologue: done-at-depot check, state features, se = sf@state_w+b,
// se_dyn = se@dyn_w1[H:], seCtx = se@ptr_w1[2H:] + ptr_b1. Block per batch.
__global__ void step_state(const float* __restrict__ caps,
                           const float* __restrict__ dists,
                           const float* __restrict__ state_w,
                           const float* __restrict__ state_b,
                           const float* __restrict__ dyn_w1,
                           const float* __restrict__ ptr_w1,
                           const float* __restrict__ ptr_b1,
                           const unsigned char* __restrict__ visited,
                           const int* __restrict__ cur,
                           const float* __restrict__ rem,
                           int* __restrict__ doneF,
                           float* __restrict__ seDyn, float* __restrict__ seCtx,
                           int t) {
  int b = blockIdx.x, j = threadIdx.x;
  __shared__ float sf[4];
  __shared__ float se[H_];
  if (j == 0) {
    int c = cur[b];
    int all1 = 1, vc = 0;
    for (int n = 0; n < N_; ++n) {
      int vi = visited[b * N_ + n];
      vc += vi;
      if (n >= 1) all1 &= vi;
    }
    if (all1 && c == 0) doneF[b] = 1;
    float cap = caps[b];
    sf[0] = (cap - rem[b]) / cap;
    sf[1] = (float)t / (float)MAXS;
    sf[2] = (float)vc / (float)N_;
    sf[3] = dists[((size_t)b * N_ + c) * N_];
  }
  __syncthreads();
  float s = state_b[j];
#pragma unroll
  for (int kk = 0; kk < 4; ++kk) s += sf[kk] * state_w[kk * H_ + j];
  se[j] = s;
  __syncthreads();
  float sd = 0.f, sp = ptr_b1[j];
  for (int kk = 0; kk < H_; ++kk) {
    float sv = se[kk];
    sd += sv * dyn_w1[(size_t)(H_ + kk) * H_ + j];
    sp += sv * ptr_w1[(size_t)(2 * H_ + kk) * H_ + j];
  }
  seDyn[(size_t)b * H_ + j] = sd;
  seCtx[(size_t)b * H_ + j] = sp;
}

// g = mean_n(ue); ctxsum = seCtx + g @ ptr_w1[H:2H]. Block per batch.
__global__ void reduce_g(const _Float16* __restrict__ ue,
                         const float* __restrict__ ptr_w1,
                         const float* __restrict__ seCtx,
                         float* __restrict__ ctxsum) {
  int b = blockIdx.x, j = threadIdx.x;
  __shared__ float g[H_];
  float s = 0.f;
  for (int n = 0; n < N_; ++n) s += (float)ue[((size_t)b * N_ + n) * H_ + j];
  g[j] = s * (1.f / N_);
  __syncthreads();
  float acc = seCtx[(size_t)b * H_ + j];
  for (int kk = 0; kk < H_; ++kk)
    acc += g[kk] * ptr_w1[(size_t)(H_ + kk) * H_ + j];
  ctxsum[(size_t)b * H_ + j] = acc;
}

__global__ void seed_scores(float* __restrict__ scores,
                            const float* __restrict__ pb2, int n) {
  int i = blockIdx.x * 256 + threadIdx.x;
  if (i < n) scores[i] = pb2[0];
}

// Masking + log_softmax + greedy argmax + state update. Block per batch.
__global__ void decode_k(const float* __restrict__ scores,
                         const float* __restrict__ demands,
                         const float* __restrict__ caps,
                         unsigned char* __restrict__ visited,
                         int* __restrict__ cur, float* __restrict__ rem,
                         int* __restrict__ doneF, float* __restrict__ lps,
                         float* __restrict__ out, int t) {
  int b = blockIdx.x, j = threadIdx.x;
  __shared__ float sc[H_];
  __shared__ int   msk[H_];
  __shared__ float redf[H_];
  __shared__ int   redi[H_];
  int   c = cur[b];
  float r = rem[b];
  int done = doneF[b];
  float s = NEGV;
  int   m = 1;
  if (j < N_) {
    s = scores[(size_t)b * N_ + j];
    m = (visited[b * N_ + j] != 0) || (demands[(size_t)b * N_ + j] > r);
    if (j == 0 && c == 0) m = 1;
  }
  msk[j] = m;
  redi[j] = m; __syncthreads();
  for (int o = 64; o > 0; o >>= 1) { if (j < o) redi[j] &= redi[j + o]; __syncthreads(); }
  int allm = redi[0]; __syncthreads();
  if (allm && c != 0 && j == 0) msk[0] = 0;
  if (allm && c == 0) done = 1;
  __syncthreads();
  m = msk[j];
  float sm = (m || j >= N_) ? NEGV : s;
  sc[j] = sm;
  redf[j] = sm; __syncthreads();
  for (int o = 64; o > 0; o >>= 1) { if (j < o) redf[j] = fmaxf(redf[j], redf[j + o]); __syncthreads(); }
  float mx = redf[0]; __syncthreads();
  redf[j] = (j < N_) ? __expf(sm - mx) : 0.f; __syncthreads();
  for (int o = 64; o > 0; o >>= 1) { if (j < o) redf[j] += redf[j + o]; __syncthreads(); }
  float lse = __logf(redf[0]); __syncthreads();
  redf[j] = sm; redi[j] = j; __syncthreads();
  for (int o = 64; o > 0; o >>= 1) {
    if (j < o) {
      float v2 = redf[j + o]; int i2 = redi[j + o];
      if (v2 > redf[j] || (v2 == redf[j] && i2 < redi[j])) { redf[j] = v2; redi[j] = i2; }
    }
    __syncthreads();
  }
  int action = done ? 0 : redi[0];
  if (j == 0) {
    float slp = done ? 0.f : (sc[action] - mx - lse);
    lps[b] += slp;
    int to_depot = (action == 0);
    if (!done) {
      rem[b] = to_depot ? caps[b] : r - demands[(size_t)b * N_ + action];
      if (!to_depot) visited[b * N_ + action] = 1;
      cur[b] = action;
    }
    doneF[b] = done;
    out[(size_t)b * MAXS + t] = (float)action;
    if (t == MAXS - 1) out[(size_t)B_ * MAXS + b] = lps[b];
  }
}

// ---------------------------------------------------------------------------
// Host side. Input order = setup_inputs() dict insertion order, params nested
// dicts flattened in insertion order (64 arrays total).
// ---------------------------------------------------------------------------
extern "C" void kernel_launch(void* const* d_in, const int* in_sizes, int n_in,
                              void* d_out, int out_size, void* d_ws, size_t ws_size,
                              hipStream_t stream) {
  (void)in_sizes; (void)n_in; (void)out_size; (void)ws_size;
  const float* coords   = (const float*)d_in[0];
  const float* demands  = (const float*)d_in[1];
  const float* caps     = (const float*)d_in[2];
  const float* dists    = (const float*)d_in[3];
  const float* embed_w  = (const float*)d_in[4];
  const float* embed_b  = (const float*)d_in[5];
  const int LBASE = 6, LSTRIDE = 16;
  const float* state_w = (const float*)d_in[54];
  const float* state_b = (const float*)d_in[55];
  const float* dyn_w1  = (const float*)d_in[56];
  const float* dyn_b1  = (const float*)d_in[57];
  const float* dyn_w2  = (const float*)d_in[58];
  const float* dyn_b2  = (const float*)d_in[59];
  const float* ptr_w1  = (const float*)d_in[60];
  const float* ptr_b1  = (const float*)d_in[61];
  const float* ptr_w2  = (const float*)d_in[62];
  const float* ptr_b2  = (const float*)d_in[63];
  float* out = (float*)d_out;

  // ---- workspace carving (aligned to 256 B) ----
  char* cur_ws = (char*)d_ws;
  auto carve = [&](size_t bytes) -> void* {
    void* p = (void*)cur_ws;
    cur_ws += (bytes + 255) & ~size_t(255);
    return p;
  };
  const size_t MH = (size_t)M_ * H_;
  float*     xA      = (float*)carve(MH * 4);         // fp32 activations / emb
  float*     xB      = (float*)carve(MH * 4);
  float*     pf      = (float*)carve(MH * 4);         // GEMM fp32 out (O/FFN2)
  float*     embDyn  = (float*)carve(MH * 4);         // emb@dyn_w1[:H]+b1
  _Float16*  hA      = (_Float16*)carve(MH * 2);      // fp16 twins
  _Float16*  hB      = (_Float16*)carve(MH * 2);
  _Float16*  qh      = (_Float16*)carve(MH * 2);
  _Float16*  kh      = (_Float16*)carve(MH * 2);
  _Float16*  vh      = (_Float16*)carve(MH * 2);
  _Float16*  ah      = (_Float16*)carve(MH * 2);
  _Float16*  ueh     = (_Float16*)carve(MH * 2);
  _Float16*  fh      = (_Float16*)carve((size_t)M_ * DFF_ * 2);
  // fp16 transposed weights
  _Float16* wT[NL_][6];
  for (int l = 0; l < NL_; ++l) {
    for (int wti = 0; wti < 4; ++wti) wT[l][wti] = (_Float16*)carve(H_ * H_ * 2);
    wT[l][4] = (_Float16*)carve(H_ * DFF_ * 2);       // w1^T  [DFF,H]
    wT[l][5] = (_Float16*)carve(DFF_ * H_ * 2);       // w2^T  [H,DFF]
  }
  _Float16* dynw1AT = (_Float16*)carve(H_ * H_ * 2);
  _Float16* dynw2T  = (_Float16*)carve(H_ * H_ * 2);
  _Float16* ptrAT   = (_Float16*)carve(H_ * H_ * 2);
  // decoder state / per-step vectors
  float* seDyn  = (float*)carve((size_t)B_ * H_ * 4);
  float* seCtx  = (float*)carve((size_t)B_ * H_ * 4);
  float* ctxsum = (float*)carve((size_t)B_ * H_ * 4);
  float* scores = (float*)carve((size_t)M_ * 4);
  unsigned char* visited = (unsigned char*)carve(B_ * N_);
  int*   curI  = (int*)carve(B_ * 4);
  float* remF  = (float*)carve(B_ * 4);
  int*   doneF = (int*)carve(B_ * 4);
  float* lpsF  = (float*)carve(B_ * 4);

  const int CT = 256;
  auto cgrid = [](int n, int b) { return (n + b - 1) / b; };

  // ---- weight conversion (fp32 -> fp16 transposed) ----
  const int wmap[4] = {0, 2, 4, 6};  // wq wk wv wo offsets in layer dict
  for (int l = 0; l < NL_; ++l) {
    const int base = LBASE + l * LSTRIDE;
    for (int wti = 0; wti < 4; ++wti)
      conv_w_f16t<<<cgrid(H_ * H_, CT), CT, 0, stream>>>(
          (const float*)d_in[base + wmap[wti]], wT[l][wti], H_, H_, 0);
    conv_w_f16t<<<cgrid(H_ * DFF_, CT), CT, 0, stream>>>(
        (const float*)d_in[base + 10], wT[l][4], H_, DFF_, 0);
    conv_w_f16t<<<cgrid(DFF_ * H_, CT), CT, 0, stream>>>(
        (const float*)d_in[base + 12], wT[l][5], DFF_, H_, 0);
  }
  conv_w_f16t<<<cgrid(H_ * H_, CT), CT, 0, stream>>>(dyn_w1, dynw1AT, H_, H_, 0);
  conv_w_f16t<<<cgrid(H_ * H_, CT), CT, 0, stream>>>(dyn_w2, dynw2T, H_, H_, 0);
  conv_w_f16t<<<cgrid(H_ * H_, CT), CT, 0, stream>>>(ptr_w1, ptrAT, H_, H_, 0);

  // ---- encoder ----
  embed_k<<<M_, H_, 0, stream>>>(coords, demands, embed_w, embed_b, xA, hA);
  const int G128 = M_ / 16;        // 6400 blocks: 1 group of 8 col tiles
  const int G256 = (M_ / 16) * 2;  // 12800 blocks: 2 groups (Nc=256)
  for (int l = 0; l < NL_; ++l) {
    const int base = LBASE + l * LSTRIDE;
    const float* bq   = (const float*)d_in[base + 1];
    const float* bk   = (const float*)d_in[base + 3];
    const float* bv   = (const float*)d_in[base + 5];
    const float* bo   = (const float*)d_in[base + 7];
    const float* l1s  = (const float*)d_in[base + 8];
    const float* l1b  = (const float*)d_in[base + 9];
    const float* b1   = (const float*)d_in[base + 11];
    const float* b2   = (const float*)d_in[base + 13];
    const float* l2s  = (const float*)d_in[base + 14];
    const float* l2b  = (const float*)d_in[base + 15];
    gemm_f16<0, 1><<<G128, 256, 0, stream>>>(hA, wT[l][0], bq, nullptr, qh, M_, H_, H_);
    gemm_f16<0, 1><<<G128, 256, 0, stream>>>(hA, wT[l][1], bk, nullptr, kh, M_, H_, H_);
    gemm_f16<0, 1><<<G128, 256, 0, stream>>>(hA, wT[l][2], bv, nullptr, vh, M_, H_, H_);
    attn_k<<<B_ * NH_, 128, 0, stream>>>(qh, kh, vh, ah);
    gemm_f16<0, 0><<<G128, 256, 0, stream>>>(ah, wT[l][3], bo, pf, nullptr, M_, H_, H_);
    add_ln<<<M_, H_, 0, stream>>>(pf, xA, l1s, l1b, xB, hB);
    gemm_f16<1, 1><<<G256, 256, 0, stream>>>(hB, wT[l][4], b1, nullptr, fh, M_, H_, DFF_);
    gemm_f16<0, 0><<<G128, 256, 0, stream>>>(fh, wT[l][5], b2, pf, nullptr, M_, DFF_, H_);
    add_ln<<<M_, H_, 0, stream>>>(pf, xB, l2s, l2b, xA, hA);
  }
  // step-invariant: embDyn = emb @ dyn_w1[:H] + dyn_b1
  gemm_f16<0, 0><<<G128, 256, 0, stream>>>(hA, dynw1AT, dyn_b1, embDyn, nullptr, M_, H_, H_);

  // ---- decoder ----
  init_state<<<cgrid(B_ * N_, CT), CT, 0, stream>>>(visited, curI, remF, doneF, lpsF, caps);
  for (int t = 0; t < MAXS; ++t) {
    step_state<<<B_, H_, 0, stream>>>(caps, dists, state_w, state_b, dyn_w1,
                                      ptr_w1, ptr_b1, visited, curI, remF,
                                      doneF, seDyn, seCtx, t);
    gemm1_step<<<M_ / 16, 256, 0, stream>>>(embDyn, seDyn, dynw2T, dyn_b2, xA, ueh);
    reduce_g<<<B_, H_, 0, stream>>>(ueh, ptr_w1, seCtx, ctxsum);
    seed_scores<<<cgrid(M_, CT), CT, 0, stream>>>(scores, ptr_b2, M_);
    gemm2_step<<<M_ / 16, 256, 0, stream>>>(ueh, ptrAT, ctxsum, ptr_w2, scores);
    decode_k<<<B_, H_, 0, stream>>>(scores, demands, caps, visited, curI, remF,
                                    doneF, lpsF, out, t);
  }
}
